// SelectiveSSM_16716012716040
// MI455X (gfx1250) — matrix operand order repaired and checked by reading, not compiled, not run on hardware
//
#include <hip/hip_runtime.h>
#include <hip/hip_bf16.h>

// ---------------------------------------------------------------------------
// Selective SSM (Mamba-style) for MI455X / gfx1250, wave32 + WMMA f16.
//   cvt:    f32 -> f16 input; f32 -> f16 *transposed* W_in / W_out
//   GEMM1:  xz = X @ W_in        (WMMA f32_16x16x32_f16, 128x128 tiles,
//                                 double-buffered LDS, 1 barrier / k-step)
//   conv:   depthwise conv4 + SiLU
//   xdbl:   conv @ W_x (N=33) -> delta/softplus, dA, dBx, C
//   scan:   parallel chunked affine scan per (batch, state-dim)
//   gate:   (ys + conv*D) * silu(gated)  -> f16
//   GEMM2:  out = gated_out @ W_out (WMMA, f32 result)
// ---------------------------------------------------------------------------

typedef __attribute__((ext_vector_type(16))) _Float16 v16h;
typedef __attribute__((ext_vector_type(8)))  _Float16 v8h;
typedef __attribute__((ext_vector_type(4)))  _Float16 v4h;
typedef __attribute__((ext_vector_type(8)))  float    v8f;

#define D_MODEL 1024
#define D_STATE 16
#define D_CONV  4
#define D_INNER 2048
#define BATCH   2
#define SEQ     4096
#define NTOK    (BATCH * SEQ)   // 8192

__device__ __forceinline__ float silu_f(float x) {
    return x / (1.0f + __expf(-x));
}

// ---------------------------------------------------------------------------
// f32 -> f16 conversion, 4 elements per thread
// ---------------------------------------------------------------------------
__global__ void cvt_f32_f16_v4(const float* __restrict__ s,
                               _Float16* __restrict__ d, int n4) {
    int i = blockIdx.x * blockDim.x + threadIdx.x;
    if (i < n4) {
        float4 x = ((const float4*)s)[i];
        v4h h = {(_Float16)x.x, (_Float16)x.y, (_Float16)x.z, (_Float16)x.w};
        ((v4h*)d)[i] = h;
    }
}

// ---------------------------------------------------------------------------
// f32 (K x N, row-major) -> f16 transposed (N x K).  32x32 LDS tiles.
// grid = (N/32, K/32), block = 256 (32x8).
// ---------------------------------------------------------------------------
__global__ __launch_bounds__(256) void cvt_transpose_f32_f16(
    const float* __restrict__ s, _Float16* __restrict__ d, int K, int N) {
    __shared__ float tile[32][33];
    int n0 = blockIdx.x * 32;
    int k0 = blockIdx.y * 32;
    int tx = threadIdx.x & 31;
    int ty = threadIdx.x >> 5;   // 0..7
#pragma unroll
    for (int i = 0; i < 32; i += 8)
        tile[ty + i][tx] = s[(size_t)(k0 + ty + i) * N + n0 + tx];
    __syncthreads();
#pragma unroll
    for (int i = 0; i < 32; i += 8)
        d[(size_t)(n0 + ty + i) * K + k0 + tx] = (_Float16)tile[tx][ty + i];
}

// ---------------------------------------------------------------------------
// Tiled WMMA GEMM: C[MxN] = A[MxK] * B[KxN], with B supplied TRANSPOSED:
// BT is N x K row-major f16.  128x128 tile / 256-thread block (8 waves);
// each wave computes 32x64 (2x4 WMMA accumulators).  Double-buffered LDS.
// ---------------------------------------------------------------------------
template <bool OUT_F16>
__global__ __launch_bounds__(256) void gemm_wmma_f16(
    const _Float16* __restrict__ A,    // M x K
    const _Float16* __restrict__ BT,   // N x K
    void* __restrict__ Cout,           // M x N
    int M, int N, int K) {
    __shared__ _Float16 As[2][128][40];   // [buf][m][k], 80B rows
    __shared__ _Float16 Bs[2][128][40];   // [buf][n][k]

    const int tid  = threadIdx.x;
    const int lane = tid & 31;
    const int wave = tid >> 5;
    const int wm   = wave >> 1;        // 0..3
    const int wn   = wave & 1;         // 0..1
    const int half = lane >> 4;        // 0/1
    const int ml   = lane & 15;

    const int n0 = blockIdx.x * 128;
    const int m0 = blockIdx.y * 128;

    // tile-load assignment (identical for A and BT): 2 threads per row
    const int lr = tid >> 1;           // row 0..127 (m for A, n for BT)
    const int lk = (tid & 1) * 16;     // k-offset 0/16

    const _Float16* aptr = A  + (size_t)(m0 + lr) * K + lk;
    const _Float16* bptr = BT + (size_t)(n0 + lr) * K + lk;

    v8f acc[2][4] = {};

    // prologue: stage k-tile 0 into buffer 0
    {
        v16h av = *(const v16h*)(aptr);
        v16h bv = *(const v16h*)(bptr);
        *(v8h*)&As[0][lr][lk]     = __builtin_shufflevector(av, av, 0, 1, 2, 3, 4, 5, 6, 7);
        *(v8h*)&As[0][lr][lk + 8] = __builtin_shufflevector(av, av, 8, 9, 10, 11, 12, 13, 14, 15);
        *(v8h*)&Bs[0][lr][lk]     = __builtin_shufflevector(bv, bv, 0, 1, 2, 3, 4, 5, 6, 7);
        *(v8h*)&Bs[0][lr][lk + 8] = __builtin_shufflevector(bv, bv, 8, 9, 10, 11, 12, 13, 14, 15);
    }
    __syncthreads();

    const int nk = K >> 5;   // K/32 steps
    for (int it = 0; it < nk; ++it) {
        const int p = it & 1;
        const bool has_next = (it + 1 < nk);

        // issue next tile's global loads early (overlap with WMMA below)
        v16h nav, nbv;
        if (has_next) {
            nav = *(const v16h*)(aptr + (size_t)(it + 1) * 32);
            nbv = *(const v16h*)(bptr + (size_t)(it + 1) * 32);
        }

        // fragment loads from buffer p (ISA operand layouts)
        v16h afr[2], bfr[4];
#pragma unroll
        for (int tm = 0; tm < 2; ++tm) {
            int row = wm * 32 + tm * 16 + ml;
            v8h lo = *(const v8h*)&As[p][row][half * 8];
            v8h hi = *(const v8h*)&As[p][row][16 + half * 8];
            afr[tm] = __builtin_shufflevector(lo, hi, 0, 1, 2, 3, 4, 5, 6, 7,
                                              8, 9, 10, 11, 12, 13, 14, 15);
        }
#pragma unroll
        for (int tn = 0; tn < 4; ++tn) {
            int col = wn * 64 + tn * 16 + ml;
            v8h b0 = *(const v8h*)&Bs[p][col][half * 8];
            v8h b1 = *(const v8h*)&Bs[p][col][16 + half * 8];
            bfr[tn] = __builtin_shufflevector(b0, b1, 0, 1, 2, 3, 4, 5, 6, 7,
                                              8, 9, 10, 11, 12, 13, 14, 15);
        }
#pragma unroll
        for (int tm = 0; tm < 2; ++tm)
#pragma unroll
            for (int tn = 0; tn < 4; ++tn)
                acc[tm][tn] = __builtin_amdgcn_wmma_f32_16x16x32_f16(
                    false, afr[tm], false, bfr[tn], (short)0, acc[tm][tn],
                    false, false);

        // stage next tile into the other buffer
        if (has_next) {
            const int q = 1 - p;
            *(v8h*)&As[q][lr][lk]     = __builtin_shufflevector(nav, nav, 0, 1, 2, 3, 4, 5, 6, 7);
            *(v8h*)&As[q][lr][lk + 8] = __builtin_shufflevector(nav, nav, 8, 9, 10, 11, 12, 13, 14, 15);
            *(v8h*)&Bs[q][lr][lk]     = __builtin_shufflevector(nbv, nbv, 0, 1, 2, 3, 4, 5, 6, 7);
            *(v8h*)&Bs[q][lr][lk + 8] = __builtin_shufflevector(nbv, nbv, 8, 9, 10, 11, 12, 13, 14, 15);
        }
        __syncthreads();
    }

    // store D: VGPR v -> m = v + half*8 (within 16x16 tile), n = ml
#pragma unroll
    for (int tm = 0; tm < 2; ++tm)
#pragma unroll
        for (int tn = 0; tn < 4; ++tn) {
            int gn = n0 + wn * 64 + tn * 16 + ml;
#pragma unroll
            for (int v = 0; v < 8; ++v) {
                int gm = m0 + wm * 32 + tm * 16 + half * 8 + v;
                float val = acc[tm][tn][v];
                if (OUT_F16)
                    ((_Float16*)Cout)[(size_t)gm * N + gn] = (_Float16)val;
                else
                    ((float*)Cout)[(size_t)gm * N + gn] = val;
            }
        }
}

// ---------------------------------------------------------------------------
// Depthwise causal conv (k=4) + bias + SiLU.  grid = (2048/256, NTOK)
// ---------------------------------------------------------------------------
__global__ __launch_bounds__(256) void conv_silu_kernel(
    const _Float16* __restrict__ xz16,   // [NTOK][4096], first 2048 = conv_in
    const float* __restrict__ cw,        // [2048][1][4]
    const float* __restrict__ cb,        // [2048]
    _Float16* __restrict__ conv16) {     // [NTOK][2048]
    int c = blockIdx.x * 256 + threadIdx.x;
    int token = blockIdx.y;
    int b = token >> 12;
    int t = token & (SEQ - 1);
    float acc = cb[c];
#pragma unroll
    for (int j = 0; j < D_CONV; ++j) {
        int tt = t - (D_CONV - 1) + j;
        if (tt >= 0)
            acc += cw[c * D_CONV + j] *
                   (float)xz16[(size_t)(b * SEQ + tt) * 4096 + c];
    }
    conv16[(size_t)token * D_INNER + c] = (_Float16)silu_f(acc);
}

// ---------------------------------------------------------------------------
// x_dbl = conv @ W_x (2048 x 33), then delta/softplus, dA, dBx, C.
// One 256-thread block per token; wave shuffle reduce + LDS atomics.
// ---------------------------------------------------------------------------
__global__ __launch_bounds__(256) void xdbl_kernel(
    const _Float16* __restrict__ conv16,
    const float* __restrict__ Wx,        // [2048][33]
    const float* __restrict__ A_log,     // [16]
    float* __restrict__ dA,              // [NTOK][16]
    float* __restrict__ dBx,             // [NTOK][16]
    float* __restrict__ Cc) {            // [NTOK][16]
    __shared__ float xd[33];
    const int token = blockIdx.x;
    const int tid = threadIdx.x;
    if (tid < 33) xd[tid] = 0.0f;
    __syncthreads();

    float p[33];
#pragma unroll
    for (int c = 0; c < 33; ++c) p[c] = 0.0f;

    for (int ch = tid; ch < D_INNER; ch += 256) {
        float cv = (float)conv16[(size_t)token * D_INNER + ch];
        const float* wr = Wx + ch * 33;
#pragma unroll
        for (int c = 0; c < 33; ++c) p[c] += cv * wr[c];
    }
#pragma unroll
    for (int c = 0; c < 33; ++c) {
        float v = p[c];
        for (int off = 16; off > 0; off >>= 1) v += __shfl_down(v, off, 32);
        if ((tid & 31) == 0) atomicAdd(&xd[c], v);
    }
    __syncthreads();

    if (tid < D_STATE) {
        float x0 = xd[0];
        float delta = (x0 > 20.0f) ? x0 : log1pf(__expf(x0));
        float Aj = -__expf(A_log[tid]);
        float cv = (float)conv16[(size_t)token * D_INNER + tid];
        size_t o = (size_t)token * D_STATE + tid;
        dA[o]  = __expf(delta * Aj);
        dBx[o] = delta * xd[1 + tid] * cv;
        Cc[o]  = xd[17 + tid];
    }
}

// ---------------------------------------------------------------------------
// Parallel scan: h_t = dA_t*h_{t-1} + dBx_t ; contribution h_t * C_t[j].
// One block per (batch, state-dim) = 32 blocks; 256 threads x 16-step chunks.
// Writes per-dim partials (deterministic reduce afterwards).
// ---------------------------------------------------------------------------
__global__ __launch_bounds__(256) void scan_kernel(
    const float* __restrict__ dA,
    const float* __restrict__ dBx,
    const float* __restrict__ Cc,
    float* __restrict__ ysp) {           // [16][NTOK]
    __shared__ float sA[256], sB[256];
    const int b = blockIdx.x >> 4;
    const int j = blockIdx.x & 15;
    const int t = threadIdx.x;
    const int s0 = t * 16;
    const size_t base = ((size_t)(b * SEQ) + s0) * D_STATE + j;

    float av[16], xv[16];
#pragma unroll
    for (int i = 0; i < 16; ++i) {
        av[i] = dA[base + (size_t)i * D_STATE];
        xv[i] = dBx[base + (size_t)i * D_STATE];
    }
    // fold chunk into affine monoid: h_out = Ag*h_in + Bg
    float Ag = 1.0f, Bg = 0.0f;
#pragma unroll
    for (int i = 0; i < 16; ++i) {
        Ag = av[i] * Ag;
        Bg = av[i] * Bg + xv[i];
    }
    sA[t] = Ag;
    sB[t] = Bg;
    __syncthreads();
    // Hillis-Steele inclusive scan over chunk monoids
    for (int off = 1; off < 256; off <<= 1) {
        float pA = 0.0f, pB = 0.0f;
        bool act = (t >= off);
        if (act) { pA = sA[t - off]; pB = sB[t - off]; }
        __syncthreads();
        if (act) {
            sB[t] = sA[t] * pB + sB[t];   // uses pre-update sA[t]
            sA[t] = sA[t] * pA;
        }
        __syncthreads();
    }
    float h = (t == 0) ? 0.0f : sB[t - 1];   // h entering this chunk (h0 = 0)
#pragma unroll
    for (int i = 0; i < 16; ++i) {
        h = av[i] * h + xv[i];
        float cc = Cc[base + (size_t)i * D_STATE];
        ysp[(size_t)j * NTOK + b * SEQ + s0 + i] = h * cc;
    }
}

__global__ void ys_reduce_kernel(const float* __restrict__ ysp,
                                 float* __restrict__ ys) {
    int i = blockIdx.x * blockDim.x + threadIdx.x;
    if (i < NTOK) {
        float s = 0.0f;
#pragma unroll
        for (int j = 0; j < D_STATE; ++j) s += ysp[(size_t)j * NTOK + i];
        ys[i] = s;
    }
}

// ---------------------------------------------------------------------------
// gated_out = (ys + conv*D) * silu(gated)   -> f16.  grid = (8, NTOK)
// ---------------------------------------------------------------------------
__global__ __launch_bounds__(256) void gate_kernel(
    const _Float16* __restrict__ conv16,
    const _Float16* __restrict__ xz16,
    const float* __restrict__ ys,
    const float* __restrict__ Dp,
    _Float16* __restrict__ go16) {
    int c = blockIdx.x * 256 + threadIdx.x;
    int token = blockIdx.y;
    float skip = ys[token] + (float)conv16[(size_t)token * D_INNER + c] * Dp[c];
    float g = (float)xz16[(size_t)token * 4096 + D_INNER + c];
    go16[(size_t)token * D_INNER + c] = (_Float16)(skip * silu_f(g));
}

// ---------------------------------------------------------------------------
extern "C" void kernel_launch(void* const* d_in, const int* in_sizes, int n_in,
                              void* d_out, int out_size, void* d_ws,
                              size_t ws_size, hipStream_t stream) {
    (void)in_sizes; (void)n_in; (void)out_size; (void)ws_size;
    const float* inX   = (const float*)d_in[0];   // (2,4096,1024)
    const float* W_in  = (const float*)d_in[1];   // (1024,4096)
    const float* convw = (const float*)d_in[2];   // (2048,1,4)
    const float* convb = (const float*)d_in[3];   // (2048,)
    const float* Wx    = (const float*)d_in[4];   // (2048,33)
    const float* A_log = (const float*)d_in[5];   // (16,)
    const float* Dp    = (const float*)d_in[6];   // (2048,)
    const float* W_out = (const float*)d_in[7];   // (2048,1024)
    float* out = (float*)d_out;                   // (2,4096,1024)
    char* ws = (char*)d_ws;

    // workspace layout (all offsets 32B-aligned)
    size_t o = 0;
    _Float16* Xf16    = (_Float16*)(ws + o); o += (size_t)NTOK * D_MODEL * 2;
    _Float16* Win16T  = (_Float16*)(ws + o); o += (size_t)D_MODEL * 4096 * 2;  // 4096 x 1024
    _Float16* Wout16T = (_Float16*)(ws + o); o += (size_t)D_INNER * D_MODEL * 2; // 1024 x 2048
    _Float16* xz16    = (_Float16*)(ws + o); o += (size_t)NTOK * 4096 * 2;
    _Float16* conv16  = (_Float16*)(ws + o); o += (size_t)NTOK * D_INNER * 2;
    _Float16* go16    = (_Float16*)(ws + o); o += (size_t)NTOK * D_INNER * 2;
    float* dA  = (float*)(ws + o); o += (size_t)NTOK * D_STATE * 4;
    float* dBx = (float*)(ws + o); o += (size_t)NTOK * D_STATE * 4;
    float* Cc  = (float*)(ws + o); o += (size_t)NTOK * D_STATE * 4;
    float* ysp = (float*)(ws + o); o += (size_t)D_STATE * NTOK * 4;
    float* ys  = (float*)(ws + o); o += (size_t)NTOK * 4;

    // f32 -> f16 conversions (weights transposed: B tiles become K-contiguous)
    {
        int n4 = NTOK * D_MODEL / 4;
        cvt_f32_f16_v4<<<(n4 + 255) / 256, 256, 0, stream>>>(inX, Xf16, n4);
        cvt_transpose_f32_f16<<<dim3(4096 / 32, D_MODEL / 32), 256, 0, stream>>>(
            W_in, Win16T, D_MODEL, 4096);
        cvt_transpose_f32_f16<<<dim3(D_MODEL / 32, D_INNER / 32), 256, 0, stream>>>(
            W_out, Wout16T, D_INNER, D_MODEL);
    }

    // GEMM1: xz = X @ W_in   (8192x1024 * 1024x4096)
    {
        dim3 grid(4096 / 128, NTOK / 128);
        gemm_wmma_f16<true><<<grid, 256, 0, stream>>>(Xf16, Win16T, (void*)xz16,
                                                      NTOK, 4096, D_MODEL);
    }

    // conv + SiLU
    conv_silu_kernel<<<dim3(D_INNER / 256, NTOK), 256, 0, stream>>>(
        xz16, convw, convb, conv16);

    // x_dbl + delta/dA/dBx/C
    xdbl_kernel<<<NTOK, 256, 0, stream>>>(conv16, Wx, A_log, dA, dBx, Cc);

    // parallel scan + deterministic reduce
    scan_kernel<<<BATCH * D_STATE, 256, 0, stream>>>(dA, dBx, Cc, ysp);
    ys_reduce_kernel<<<NTOK / 256, 256, 0, stream>>>(ysp, ys);

    // gating
    gate_kernel<<<dim3(D_INNER / 256, NTOK), 256, 0, stream>>>(conv16, xz16, ys,
                                                               Dp, go16);

    // GEMM2: out = gated_out @ W_out  (8192x2048 * 2048x1024), f32 output
    {
        dim3 grid(D_MODEL / 128, NTOK / 128);
        gemm_wmma_f16<false><<<grid, 256, 0, stream>>>(go16, Wout16T, (void*)out,
                                                       NTOK, D_MODEL, D_INNER);
    }
}
